// Deeper_EEG_VQ_VAE_84748294685324
// MI455X (gfx1250) — matrix operand order repaired and checked by reading, not compiled, not run on hardware
//
#include <hip/hip_runtime.h>
#include <hip/hip_bf16.h>

typedef _Float16 h16;
typedef __attribute__((ext_vector_type(16))) _Float16 v16h;
typedef __attribute__((ext_vector_type(8)))  _Float16 v8h;
typedef __attribute__((ext_vector_type(8)))  float    v8f;

#define WMMA(a, b, c) __builtin_amdgcn_wmma_f32_16x16x32_f16( \
    false, (a), false, (b), (short)0, (c), false, false)

// Problem sizes (compile-time constants from the reference)
constexpr int BATCH = 65536;
constexpr int IN    = 256;
constexpr int HID   = 256;
constexpr int H2    = 128;
constexpr int EMB   = 64;
constexpr int KCODE = 4096;
constexpr int WAVES = 8;   // waves per workgroup
constexpr int TILES = 2;   // 16-row sub-tiles per wave -> 32 rows/wave

// ---------------- workspace layout (bytes) ----------------
// W1h  f16 [256][256]   @ 0        (131072)
// W2h  f16 [128][256]   @ 131072   ( 65536)
// W3h  f16 [ 64][128]   @ 196608   ( 16384)
// CBh  f16 [4096][64]   @ 212992   (524288)
// S    f32 [256]        @ 737280   (  1024)   gamma * rsqrt(var+eps)
// T    f32 [256]        @ 738304   (  1024)   (b1 - mean)*S + beta
// EN   f32 [4096]       @ 739328   ( 16384)   ||e_k||^2

__global__ __launch_bounds__(256) void vq_prep(
    const float* __restrict__ W1, const float* __restrict__ W2,
    const float* __restrict__ W3, const float* __restrict__ cb,
    const float* __restrict__ b1, const float* __restrict__ gamma,
    const float* __restrict__ beta, const float* __restrict__ mean,
    const float* __restrict__ var,
    h16* __restrict__ W1h, h16* __restrict__ W2h, h16* __restrict__ W3h,
    h16* __restrict__ CBh, float* __restrict__ S, float* __restrict__ T,
    float* __restrict__ EN)
{
    const int i = blockIdx.x * 256 + threadIdx.x;
    if (i < HID * IN)   W1h[i] = (h16)W1[i];
    if (i < H2 * HID)   W2h[i] = (h16)W2[i];
    if (i < EMB * H2)   W3h[i] = (h16)W3[i];
    if (i < KCODE*EMB)  CBh[i] = (h16)cb[i];
    if (i < HID) {
        float s = gamma[i] * rsqrtf(var[i] + 1e-5f);
        S[i] = s;
        T[i] = (b1[i] - mean[i]) * s + beta[i];
    }
    if (i < KCODE) {
        float acc = 0.f;
        const float* row = cb + (size_t)i * EMB;
        #pragma unroll 8
        for (int j = 0; j < EMB; ++j) acc = fmaf(row[j], row[j], acc);
        EN[i] = acc;
    }
}

__global__ __launch_bounds__(256) void vq_fused(
    const float* __restrict__ x,
    const h16* __restrict__ W1h, const h16* __restrict__ W2h,
    const h16* __restrict__ W3h, const h16* __restrict__ CBh,
    const float* __restrict__ S, const float* __restrict__ T,
    const float* __restrict__ b2, const float* __restrict__ b3,
    const float* __restrict__ EN, int* __restrict__ out)
{
    // per-wave private LDS slices; h1/h2 scratch reused across the two
    // 16-row sub-tiles (DS ops are in-order per wave -> WAR reuse is safe).
    __shared__ __align__(16) h16 h1s[WAVES][16][HID];        // 64 KB
    __shared__ __align__(16) h16 h2s[WAVES][16][H2];         // 32 KB
    __shared__ __align__(16) h16 zs [WAVES][TILES * 16][EMB];// 32 KB

    const int w     = threadIdx.x >> 5;
    const int lane  = threadIdx.x & 31;
    const int nIdx  = lane & 15;   // B/D column within a 16-tile
    const int mIdx  = lane & 15;   // A row
    const int kh    = lane >> 4;   // k-half selector
    const int rowBase = (blockIdx.x * WAVES + w) * (TILES * 16);

    // ============ Phases 1-3 (encoder), once per 16-row sub-tile ============
    #pragma unroll 1
    for (int t = 0; t < TILES; ++t) {
        const int rb = rowBase + t * 16;

        // ---- Phase 1: h1 = relu(BN(x @ W1^T + b1)) ----
        v16h ax[8];
        {
            const float* xr = x + (size_t)(rb + mIdx) * IN;
            #pragma unroll
            for (int kt = 0; kt < 8; ++kt) {
                const int k0 = kt * 32 + kh * 16;
                float4 p0 = *(const float4*)(xr + k0);
                float4 p1 = *(const float4*)(xr + k0 + 4);
                float4 p2 = *(const float4*)(xr + k0 + 8);
                float4 p3 = *(const float4*)(xr + k0 + 12);
                v16h a;
                a[0] =(h16)p0.x; a[1] =(h16)p0.y; a[2] =(h16)p0.z; a[3] =(h16)p0.w;
                a[4] =(h16)p1.x; a[5] =(h16)p1.y; a[6] =(h16)p1.z; a[7] =(h16)p1.w;
                a[8] =(h16)p2.x; a[9] =(h16)p2.y; a[10]=(h16)p2.z; a[11]=(h16)p2.w;
                a[12]=(h16)p3.x; a[13]=(h16)p3.y; a[14]=(h16)p3.z; a[15]=(h16)p3.w;
                ax[kt] = a;
            }
        }
        #pragma unroll 1
        for (int nt = 0; nt < 16; ++nt) {
            v8f acc = {0.f,0.f,0.f,0.f,0.f,0.f,0.f,0.f};
            const h16* wr = W1h + (size_t)(nt * 16 + nIdx) * IN;
            #pragma unroll
            for (int kt = 0; kt < 8; ++kt) {
                v16h b = *(const v16h*)(wr + kt * 32 + kh * 16);
                acc = WMMA(ax[kt], b, acc);
            }
            const int j = nt * 16 + nIdx;
            const float s = S[j], tt = T[j];
            #pragma unroll
            for (int r = 0; r < 8; ++r) {
                float h = fmaxf(fmaf(acc[r], s, tt), 0.f);
                h1s[w][r + 8 * kh][j] = (h16)h;
            }
        }

        // ---- Phase 2: h2 = relu(h1 @ W2^T + b2) ----
        v16h a2[8];
        {
            const h16* hr = &h1s[w][mIdx][0];
            #pragma unroll
            for (int kt = 0; kt < 8; ++kt) {
                const int k0 = kt * 32 + kh * 16;
                v8h lo = *(const v8h*)(hr + k0);
                v8h hi = *(const v8h*)(hr + k0 + 8);
                a2[kt] = __builtin_shufflevector(lo, hi,
                         0,1,2,3,4,5,6,7,8,9,10,11,12,13,14,15);
            }
        }
        #pragma unroll 1
        for (int nt = 0; nt < 8; ++nt) {
            v8f acc = {0.f,0.f,0.f,0.f,0.f,0.f,0.f,0.f};
            const h16* wr = W2h + (size_t)(nt * 16 + nIdx) * HID;
            #pragma unroll
            for (int kt = 0; kt < 8; ++kt) {
                v16h b = *(const v16h*)(wr + kt * 32 + kh * 16);
                acc = WMMA(a2[kt], b, acc);
            }
            const int j = nt * 16 + nIdx;
            const float bias = b2[j];
            #pragma unroll
            for (int r = 0; r < 8; ++r) {
                float h = fmaxf(acc[r] + bias, 0.f);
                h2s[w][r + 8 * kh][j] = (h16)h;
            }
        }

        // ---- Phase 3: z = tanh(h2 @ W3^T + b3) ----
        v16h a3[4];
        {
            const h16* hr = &h2s[w][mIdx][0];
            #pragma unroll
            for (int kt = 0; kt < 4; ++kt) {
                const int k0 = kt * 32 + kh * 16;
                v8h lo = *(const v8h*)(hr + k0);
                v8h hi = *(const v8h*)(hr + k0 + 8);
                a3[kt] = __builtin_shufflevector(lo, hi,
                         0,1,2,3,4,5,6,7,8,9,10,11,12,13,14,15);
            }
        }
        #pragma unroll 1
        for (int nt = 0; nt < 4; ++nt) {
            v8f acc = {0.f,0.f,0.f,0.f,0.f,0.f,0.f,0.f};
            const h16* wr = W3h + (size_t)(nt * 16 + nIdx) * H2;
            #pragma unroll
            for (int kt = 0; kt < 4; ++kt) {
                v16h b = *(const v16h*)(wr + kt * 32 + kh * 16);
                acc = WMMA(a3[kt], b, acc);
            }
            const int j = nt * 16 + nIdx;
            const float bias = b3[j];
            #pragma unroll
            for (int r = 0; r < 8; ++r) {
                float zv = tanhf(acc[r] + bias);
                zs[w][t * 16 + r + 8 * kh][j] = (h16)zv;
            }
        }
    }

    // ============ Phase 4: argmin_k ||e_k||^2 - 2 z.e_k (32 rows) ============
    v16h az[TILES][2];
    #pragma unroll
    for (int t = 0; t < TILES; ++t) {
        const h16* zr = &zs[w][t * 16 + mIdx][0];
        const int k0 = kh * 16;
        v8h lo = *(const v8h*)(zr + k0);
        v8h hi = *(const v8h*)(zr + k0 + 8);
        az[t][0] = __builtin_shufflevector(lo, hi,
                   0,1,2,3,4,5,6,7,8,9,10,11,12,13,14,15);
        lo = *(const v8h*)(zr + 32 + k0);
        hi = *(const v8h*)(zr + 32 + k0 + 8);
        az[t][1] = __builtin_shufflevector(lo, hi,
                   0,1,2,3,4,5,6,7,8,9,10,11,12,13,14,15);
    }

    float bestv[TILES][8];
    int   besti[TILES][8];
    #pragma unroll
    for (int t = 0; t < TILES; ++t)
        #pragma unroll
        for (int r = 0; r < 8; ++r) { bestv[t][r] = 3.402823466e38f; besti[t][r] = 0; }

    #pragma unroll 1
    for (int ct = 0; ct < KCODE / 16; ++ct) {
        const int code = ct * 16 + nIdx;
        const h16* cr = CBh + (size_t)code * EMB;
        v16h b0 = *(const v16h*)(cr + kh * 16);        // k = 0..31
        v16h b1 = *(const v16h*)(cr + 32 + kh * 16);   // k = 32..63
        const float en = EN[code];
        #pragma unroll
        for (int t = 0; t < TILES; ++t) {
            v8f acc = {0.f,0.f,0.f,0.f,0.f,0.f,0.f,0.f};
            acc = WMMA(az[t][0], b0, acc);
            acc = WMMA(az[t][1], b1, acc);
            #pragma unroll
            for (int r = 0; r < 8; ++r) {
                float val = fmaf(-2.f, acc[r], en);   // ||e||^2 - 2 z.e
                if (val < bestv[t][r]) { bestv[t][r] = val; besti[t][r] = code; }
            }
        }
    }

    // reduce across the 16 lanes of each half (halves hold disjoint rows m)
    #pragma unroll
    for (int t = 0; t < TILES; ++t) {
        #pragma unroll
        for (int r = 0; r < 8; ++r) {
            #pragma unroll
            for (int off = 1; off < 16; off <<= 1) {
                float ov = __shfl_xor(bestv[t][r], off, 32);
                int   oi = __shfl_xor(besti[t][r], off, 32);
                if (ov < bestv[t][r] || (ov == bestv[t][r] && oi < besti[t][r])) {
                    bestv[t][r] = ov; besti[t][r] = oi;
                }
            }
        }
        if (nIdx == 0) {
            #pragma unroll
            for (int r = 0; r < 8; ++r)
                out[rowBase + t * 16 + r + 8 * kh] = besti[t][r];
        }
    }
}

extern "C" void kernel_launch(void* const* d_in, const int* in_sizes, int n_in,
                              void* d_out, int out_size, void* d_ws, size_t ws_size,
                              hipStream_t stream)
{
    (void)in_sizes; (void)n_in; (void)out_size; (void)ws_size;
    const float* x     = (const float*)d_in[0];
    const float* W1    = (const float*)d_in[1];
    const float* b1    = (const float*)d_in[2];
    const float* gamma = (const float*)d_in[3];
    const float* beta  = (const float*)d_in[4];
    const float* mean  = (const float*)d_in[5];
    const float* var   = (const float*)d_in[6];
    const float* W2    = (const float*)d_in[7];
    const float* b2    = (const float*)d_in[8];
    const float* W3    = (const float*)d_in[9];
    const float* b3    = (const float*)d_in[10];
    const float* cb    = (const float*)d_in[11];

    char* ws = (char*)d_ws;
    h16*   W1h = (h16*)(ws + 0);
    h16*   W2h = (h16*)(ws + 131072);
    h16*   W3h = (h16*)(ws + 196608);
    h16*   CBh = (h16*)(ws + 212992);
    float* S   = (float*)(ws + 737280);
    float* T   = (float*)(ws + 738304);
    float* EN  = (float*)(ws + 739328);

    vq_prep<<<(KCODE * EMB) / 256, 256, 0, stream>>>(
        W1, W2, W3, cb, b1, gamma, beta, mean, var,
        W1h, W2h, W3h, CBh, S, T, EN);

    vq_fused<<<BATCH / (WAVES * TILES * 16), 256, 0, stream>>>(
        x, W1h, W2h, W3h, CBh, S, T, b2, b3, EN, (int*)d_out);
}
